// ReferenceGroupedMLP_44702019617156
// MI455X (gfx1250) — compile-verified, matrix operand order checked by reading, not run on hardware
//
#include <hip/hip_runtime.h>
#include <stdint.h>

// Grouped SwiGLU MLP for MI455X (gfx1250, wave32, WMMA).
// G=8 groups, D=1024, H=2048, O=1024, 65536 rows (8192/group), row r -> (b=r/G, g=r%G).
//
// Strategy:
//   Kernel A: convert Wg/Wu/Wd and x fp32 -> f16 into d_ws (memory-bound pre-pass,
//             ~0.7 GB of traffic ≈ 30 us at 23.3 TB/s; the GEMM is matrix-core-bound).
//   Kernel B (per 16-row x group tile):
//     phase 0: async-stage f16 x tile into LDS via global_load_async_to_lds_b128
//              (ASYNCcnt / s_wait_asynccnt), zero VALU.
//     phase 1: hidden = silu(x Wg^T)*(x Wu^T) via v_wmma_f32_16x16x32_f16,
//              4 gate + 4 up N-tiles share each A fragment; f16 hidden -> LDS.
//     phase 2: out = hidden Wd^T, 4 N-tiles per A fragment.

#define G_  8
#define D_  1024
#define H_  2048
#define O_  1024
#define BM  16

typedef __attribute__((ext_vector_type(16))) _Float16 v16h;
typedef __attribute__((ext_vector_type(8)))  _Float16 h8;
typedef __attribute__((ext_vector_type(4)))  _Float16 h4;
typedef __attribute__((ext_vector_type(8)))  float    v8f;
typedef __attribute__((ext_vector_type(4)))  float    f4;

static constexpr size_t WG_ELEMS = (size_t)G_ * H_ * D_;   // 16,777,216 (Wg, Wu each)
static constexpr size_t WD_ELEMS = (size_t)G_ * O_ * H_;   // 16,777,216
static constexpr size_t X_ELEMS  = (size_t)65536 * D_;     // 67,108,864
static constexpr int    SMEM_BYTES = (BM * D_ + BM * H_) * (int)sizeof(_Float16); // 96 KB

__device__ __forceinline__ v16h make_frag(h8 lo, h8 hi) {
  v16h r;
  #pragma unroll
  for (int i = 0; i < 8; ++i) { r[i] = lo[i]; r[i + 8] = hi[i]; }
  return r;
}

// f16 fragment: 8 halves at p, 8 halves at p+16 (the two K-runs of the
// CDNA5 16-bit A/B VGPR layout). Two 16-byte loads, no conversion.
__device__ __forceinline__ v16h load_frag_f16(const _Float16* __restrict__ p) {
  return make_frag(*(const h8*)p, *(const h8*)(p + 16));
}

// ---------------- Kernel A: bulk fp32 -> f16 conversion ----------------
__global__ __launch_bounds__(256)
void cvt_f32_f16_kernel(const float* __restrict__ src, _Float16* __restrict__ dst,
                        long long n4) {
  long long i = (long long)blockIdx.x * blockDim.x + threadIdx.x;
  const long long stride = (long long)gridDim.x * blockDim.x;
  for (; i < n4; i += stride) {
    f4 v = *(const f4*)(src + 4 * i);
    h4 h;
    h[0] = (_Float16)v[0]; h[1] = (_Float16)v[1];
    h[2] = (_Float16)v[2]; h[3] = (_Float16)v[3];
    *(h4*)(dst + 4 * i) = h;
  }
}

// ---------------- Kernel B: fused grouped SwiGLU MLP ----------------
__global__ __launch_bounds__(256)
void grouped_swiglu_mlp_kernel(const _Float16* __restrict__ xh,
                               const _Float16* __restrict__ Wgh,
                               const _Float16* __restrict__ Wuh,
                               const _Float16* __restrict__ Wdh,
                               float* __restrict__ out)
{
  extern __shared__ _Float16 smem[];
  _Float16* xs = smem;              // [BM][D_]  x tile, f16 (32 KB)
  _Float16* hs = smem + BM * D_;    // [BM][H_]  hidden tile, f16 (64 KB)

  const int lane  = threadIdx.x & 31;
  const int wave  = threadIdx.x >> 5;
  const int l16   = lane & 15;
  const int half8 = (lane >> 4) << 3;   // 0|8: K-run base for A/B, M-offset for C/D

  const int g  = blockIdx.y;
  const int b0 = blockIdx.x * BM;

  // ---- Phase 0: async-stage x tile (f16 global -> LDS), no VGPR round-trip ----
  // 32 KB = 2048 x 16B chunks; 256 threads x 8 async b128 copies each.
  for (int i = threadIdx.x; i < BM * D_ / 8; i += 256) {
    const int m = (i * 8) / D_;
    const int d = (i * 8) % D_;
    const _Float16* gp = xh + ((size_t)(b0 + m) * G_ + g) * D_ + d;
    // Low 32 bits of a flat pointer into LDS are the LDS byte address.
    const uint32_t lds_off = (uint32_t)(uintptr_t)(&xs[m * D_ + d]);
    asm volatile("global_load_async_to_lds_b128 %0, %1, off"
                 :: "v"(lds_off), "v"(gp) : "memory");
  }
  asm volatile("s_wait_asynccnt 0" ::: "memory");
  __syncthreads();

  // ---- Phase 1: hidden = silu(x Wg^T) * (x Wu^T) ----
  // Wave covers 256 hidden cols as 4 blocks of 64 (4 N-tiles per block).
  for (int blk = 0; blk < 4; ++blk) {
    const int h0 = wave * 256 + blk * 64;
    const _Float16* wg0 = Wgh + ((size_t)g * H_ + h0 + l16) * D_;
    const _Float16* wu0 = Wuh + ((size_t)g * H_ + h0 + l16) * D_;
    __builtin_prefetch(wg0, 0, 0);      // global_prefetch_b8: pre-warm weight lines
    __builtin_prefetch(wu0, 0, 0);

    v8f cg[4] = {{}, {}, {}, {}};
    v8f cu[4] = {{}, {}, {}, {}};

    #pragma unroll 2
    for (int k0 = 0; k0 < D_; k0 += 32) {
      const v16h a = make_frag(*(const h8*)&xs[l16 * D_ + k0 + half8],
                               *(const h8*)&xs[l16 * D_ + k0 + half8 + 16]);
      #pragma unroll
      for (int t = 0; t < 4; ++t) {
        const v16h bg = load_frag_f16(wg0 + (size_t)t * 16 * D_ + k0 + half8);
        cg[t] = __builtin_amdgcn_wmma_f32_16x16x32_f16(false, a, false, bg,
                                                       (short)0, cg[t], false, false);
      }
      #pragma unroll
      for (int t = 0; t < 4; ++t) {
        const v16h bu = load_frag_f16(wu0 + (size_t)t * 16 * D_ + k0 + half8);
        cu[t] = __builtin_amdgcn_wmma_f32_16x16x32_f16(false, a, false, bu,
                                                       (short)0, cu[t], false, false);
      }
    }

    // SwiGLU epilogue (fp32) -> f16 hidden in LDS.
    // C/D layout: VGPR v holds (M = v + half8, N = l16).
    #pragma unroll
    for (int t = 0; t < 4; ++t) {
      #pragma unroll
      for (int v = 0; v < 8; ++v) {
        const int m  = v + half8;
        const float gv = cg[t][v];
        const float uv = cu[t][v];
        const float hv = (gv / (1.0f + __expf(-gv))) * uv;
        hs[m * H_ + h0 + t * 16 + l16] = (_Float16)hv;
      }
    }
  }
  __syncthreads();

  // ---- Phase 2: out = hidden Wd^T ----
  // Wave covers 128 out cols as 2 blocks of 64 (4 N-tiles per block).
  for (int blk = 0; blk < 2; ++blk) {
    const int o0 = wave * 128 + blk * 64;
    const _Float16* wd0 = Wdh + ((size_t)g * O_ + o0 + l16) * H_;
    __builtin_prefetch(wd0, 0, 0);

    v8f c[4] = {{}, {}, {}, {}};

    #pragma unroll 2
    for (int k0 = 0; k0 < H_; k0 += 32) {
      const v16h a = make_frag(*(const h8*)&hs[l16 * H_ + k0 + half8],
                               *(const h8*)&hs[l16 * H_ + k0 + half8 + 16]);
      #pragma unroll
      for (int t = 0; t < 4; ++t) {
        const v16h b = load_frag_f16(wd0 + (size_t)t * 16 * H_ + k0 + half8);
        c[t] = __builtin_amdgcn_wmma_f32_16x16x32_f16(false, a, false, b,
                                                      (short)0, c[t], false, false);
      }
    }

    #pragma unroll
    for (int t = 0; t < 4; ++t) {
      #pragma unroll
      for (int v = 0; v < 8; ++v) {
        const int m = v + half8;
        out[((size_t)(b0 + m) * G_ + g) * O_ + o0 + t * 16 + l16] = c[t][v];
      }
    }
  }
}

extern "C" void kernel_launch(void* const* d_in, const int* in_sizes, int n_in,
                              void* d_out, int out_size, void* d_ws, size_t ws_size,
                              hipStream_t stream) {
  const float* x  = (const float*)d_in[0];
  const float* Wg = (const float*)d_in[1];
  const float* Wu = (const float*)d_in[2];
  const float* Wd = (const float*)d_in[3];
  float* out = (float*)d_out;

  // f16 images in workspace: Wg(32MB) | Wu(32MB) | Wd(32MB) | x(128MB) = 224 MB
  _Float16* Wgh = (_Float16*)d_ws;
  _Float16* Wuh = Wgh + WG_ELEMS;
  _Float16* Wdh = Wuh + WG_ELEMS;
  _Float16* xh  = Wdh + WD_ELEMS;

  // Kernel A: convert weights + activations once per launch (memory-bound).
  cvt_f32_f16_kernel<<<2048, 256, 0, stream>>>(Wg, Wgh, (long long)(WG_ELEMS / 4));
  cvt_f32_f16_kernel<<<2048, 256, 0, stream>>>(Wu, Wuh, (long long)(WG_ELEMS / 4));
  cvt_f32_f16_kernel<<<2048, 256, 0, stream>>>(Wd, Wdh, (long long)(WD_ELEMS / 4));
  cvt_f32_f16_kernel<<<4096, 256, 0, stream>>>(x,  xh,  (long long)(X_ELEMS  / 4));

  const int rows = in_sizes[0] / D_;     // 65536
  const int bsz  = rows / G_;            // 8192 rows per group

  dim3 grid(bsz / BM, G_);               // 512 x 8 workgroups
  dim3 block(256);                       // 8 wave32s
  grouped_swiglu_mlp_kernel<<<grid, block, SMEM_BYTES, stream>>>(xh, Wgh, Wuh, Wdh, out);
}
